// PatchedGINConv_66340064854630
// MI455X (gfx1250) — matrix operand order repaired and checked by reading, not compiled, not run on hardware
//
#include <hip/hip_runtime.h>

typedef __attribute__((ext_vector_type(16))) _Float16 v16h;
typedef __attribute__((ext_vector_type(8)))  float    v8f;

#define N_NODES 50000
#define N_EDGES 800000
#define DIN     64
#define DHID    128
#define WAVES_PER_BLOCK 8

// ---------------------------------------------------------------------------
// Kernel 1: h = x  (self-contribution, EPS==0 -> (1+eps)*x == x)
// ---------------------------------------------------------------------------
__global__ void gin_init_kernel(const float4* __restrict__ x,
                                float4* __restrict__ h, int n4) {
  int i = blockIdx.x * blockDim.x + threadIdx.x;
  int stride = gridDim.x * blockDim.x;
  for (; i < n4; i += stride) h[i] = x[i];
}

// ---------------------------------------------------------------------------
// Kernel 2: scatter-add  h[dst] += x[src]   (16 threads per edge, 4 floats ea)
// ---------------------------------------------------------------------------
__global__ void gin_scatter_kernel(const float* __restrict__ x,
                                   const int* __restrict__ src,
                                   const int* __restrict__ dst,
                                   float* __restrict__ h) {
  int t = blockIdx.x * blockDim.x + threadIdx.x;
  int stride = gridDim.x * blockDim.x;
  const int total = N_EDGES * 16;
  for (; t < total; t += stride) {
    int e = t >> 4;
    int c = t & 15;
    int s = src[e];
    int d = dst[e];
    const float4 v = *(const float4*)(x + s * DIN + c * 4);
    float* p = h + d * DIN + c * 4;
    atomicAdd(p + 0, v.x);
    atomicAdd(p + 1, v.y);
    atomicAdd(p + 2, v.z);
    atomicAdd(p + 3, v.w);
  }
}

// ---------------------------------------------------------------------------
// Kernel 3: fused GIN MLP via WMMA.
//   h[N,64] -> relu(h@W1+b1) -> (@W2+b2) -> h[N,64]   (in place, tile-local)
// One wave32 per 16-row tile. Weights live in LDS as f16, pre-swizzled into
// v_wmma B-fragment order so each lane reads one contiguous v16h (32B).
//
// ISA layouts used (CDNA5 ISA):
//  A (16x32 f16): lane = half*16+row ; element e -> K = 16*(e/8)+half*8+(e%8)
//  B (32x16 f16): lane = khalf*16+n  ; element e -> K = 16*(e/8)+khalf*8+(e%8)
//  C/D (16x16 f32): vgpr g, lane -> M = g+8*(lane/16), N = lane%16
//
// NOTE: a zero offset is laundered through asm volatile each iteration so the
// compiler cannot LICM-hoist all 32 B-fragment LDS loads (256 VGPRs) out of
// the tile loop and spill them to scratch — they must stay ds_load_b128.
// ---------------------------------------------------------------------------
__global__ void gin_mlp_wmma_kernel(float* __restrict__ h,
                                    const float* __restrict__ W1,
                                    const float* __restrict__ b1,
                                    const float* __restrict__ W2,
                                    const float* __restrict__ b2) {
  // B-fragment-major weight staging: [frag][lane(32)][e(16)] f16
  __shared__ __align__(32) _Float16 sW1[16 * 32 * 16];  // 16 frags (kb*8+nt)
  __shared__ __align__(32) _Float16 sW2[16 * 32 * 16];  // 16 frags (kb*4+nt)
  // per-wave A-fragment restage of the 16x128 hidden tile: [kb2(4)][lane][e]
  __shared__ __align__(32) _Float16 sStage[WAVES_PER_BLOCK][4 * 32 * 16];

  const int tid = threadIdx.x;

  // --- cooperative weight load + f32->f16 + fragment swizzle ---
  for (int i = tid; i < DIN * DHID; i += blockDim.x) {   // W1: [k=64][n=128]
    int k = i / DHID, n = i % DHID;
    int kb = k >> 5, kin = k & 31, nt = n >> 4, nn = n & 15;
    int khalf = (kin >> 3) & 1, e = (kin >> 4) * 8 + (kin & 7);
    sW1[((kb * 8 + nt) * 32 + khalf * 16 + nn) * 16 + e] = (_Float16)W1[i];
  }
  for (int i = tid; i < DHID * DIN; i += blockDim.x) {   // W2: [k=128][n=64]
    int k = i / DIN, n = i % DIN;
    int kb = k >> 5, kin = k & 31, nt = n >> 4, nn = n & 15;
    int khalf = (kin >> 3) & 1, e = (kin >> 4) * 8 + (kin & 7);
    sW2[((kb * 4 + nt) * 32 + khalf * 16 + nn) * 16 + e] = (_Float16)W2[i];
  }
  __syncthreads();

  const int lane = tid & 31;
  const int wave = tid >> 5;
  const int half = lane >> 4;   // which half-wave
  const int r    = lane & 15;   // row within tile (A layout), n within tile (B/C/D)

  const int numTiles   = N_NODES / 16;                  // 3125
  const int tileStride = gridDim.x * WAVES_PER_BLOCK;
  _Float16* stg = &sStage[wave][0];

  for (int tile = blockIdx.x * WAVES_PER_BLOCK + wave; tile < numTiles;
       tile += tileStride) {
    // Opaque zero: defeats LICM of the LDS fragment loads (see note above).
    int zoff = 0;
    asm volatile("" : "+v"(zoff));

    const float* rowp = h + (tile * 16 + r) * DIN;

    // ---- A fragments for GEMM1 (K=64 -> two K=32 blocks) ----
    v16h a[2];
#pragma unroll
    for (int kb = 0; kb < 2; ++kb) {
      float tmp[16];
      *(float4*)(tmp + 0)  = *(const float4*)(rowp + kb * 32 + half * 8);
      *(float4*)(tmp + 4)  = *(const float4*)(rowp + kb * 32 + half * 8 + 4);
      *(float4*)(tmp + 8)  = *(const float4*)(rowp + kb * 32 + 16 + half * 8);
      *(float4*)(tmp + 12) = *(const float4*)(rowp + kb * 32 + 16 + half * 8 + 4);
#pragma unroll
      for (int i = 0; i < 16; ++i) a[kb][i] = (_Float16)tmp[i];
    }

    // ---- GEMM1: [16x64] @ [64x128], 8 n-tiles x 2 k-steps = 16 WMMA ----
    v8f acc1[8];
#pragma unroll
    for (int nt = 0; nt < 8; ++nt) {
      v8f c = {0.f, 0.f, 0.f, 0.f, 0.f, 0.f, 0.f, 0.f};
#pragma unroll
      for (int kb = 0; kb < 2; ++kb) {
        v16h bf = *(const v16h*)&sW1[((kb * 8 + nt) * 32 + lane) * 16 + zoff];
        c = __builtin_amdgcn_wmma_f32_16x16x32_f16(false, a[kb], false, bf,
                                                   (short)0, c, false, false);
      }
      acc1[nt] = c;
    }

    // ---- bias + ReLU, restage D-layout f32 -> A-layout f16 via LDS ----
#pragma unroll
    for (int nt = 0; nt < 8; ++nt) {
      int col = nt * 16 + r;
      float bias = b1[col];
      int kb2 = col >> 5;
      int cin = col & 31;
      int e   = (cin >> 4) * 8 + (cin & 7);  // element index in A layout
      int hh  = (cin & 15) >> 3;             // target half-wave
#pragma unroll
      for (int g = 0; g < 8; ++g) {
        int row = g + 8 * half;
        float v = acc1[nt][g] + bias;
        v = v > 0.f ? v : 0.f;
        stg[(kb2 * 32 + hh * 16 + row) * 16 + e] = (_Float16)v;
      }
    }

    // ---- A fragments for GEMM2 (K=128 -> four K=32 blocks) ----
    v16h a2[4];
#pragma unroll
    for (int kb = 0; kb < 4; ++kb)
      a2[kb] = *(const v16h*)&stg[(kb * 32 + lane) * 16 + zoff];

    // ---- GEMM2: [16x128] @ [128x64], 4 n-tiles x 4 k-steps = 16 WMMA ----
#pragma unroll
    for (int nt = 0; nt < 4; ++nt) {
      v8f c = {0.f, 0.f, 0.f, 0.f, 0.f, 0.f, 0.f, 0.f};
#pragma unroll
      for (int kb = 0; kb < 4; ++kb) {
        v16h bf = *(const v16h*)&sW2[((kb * 4 + nt) * 32 + lane) * 16 + zoff];
        c = __builtin_amdgcn_wmma_f32_16x16x32_f16(false, a2[kb], false, bf,
                                                   (short)0, c, false, false);
      }
      int col = nt * 16 + r;
      float bias = b2[col];
      float* outp = h + (tile * 16) * DIN + col;
#pragma unroll
      for (int g = 0; g < 8; ++g) {
        int row = g + 8 * half;
        outp[row * DIN] = c[g] + bias;
      }
    }
  }
}

// ---------------------------------------------------------------------------
extern "C" void kernel_launch(void* const* d_in, const int* in_sizes, int n_in,
                              void* d_out, int out_size, void* d_ws, size_t ws_size,
                              hipStream_t stream) {
  (void)in_sizes; (void)n_in; (void)out_size; (void)d_ws; (void)ws_size;
  const float* x    = (const float*)d_in[0];
  const int*   ei   = (const int*)d_in[1];     // [2,E] row-major
  const float* W1   = (const float*)d_in[2];
  const float* b1   = (const float*)d_in[3];
  const float* W2   = (const float*)d_in[4];
  const float* b2   = (const float*)d_in[5];
  float*       h    = (float*)d_out;           // used as accumulator AND output

  const int* src = ei;
  const int* dst = ei + N_EDGES;

  // 1) h = x
  const int n4 = N_NODES * DIN / 4;
  gin_init_kernel<<<2048, 256, 0, stream>>>((const float4*)x, (float4*)h, n4);

  // 2) h[dst] += x[src]
  gin_scatter_kernel<<<8192, 256, 0, stream>>>(x, src, dst, h);

  // 3) h = relu(h@W1+b1)@W2 + b2, in place (tile-local read-before-write)
  const int numTiles = N_NODES / 16;
  const int blocks = (numTiles + WAVES_PER_BLOCK - 1) / WAVES_PER_BLOCK;
  gin_mlp_wmma_kernel<<<blocks, WAVES_PER_BLOCK * 32, 0, stream>>>(h, W1, b1, W2, b2);
}
